// PCM_36086315221369
// MI455X (gfx1250) — matrix operand order repaired and checked by reading, not compiled, not run on hardware
//
#include <hip/hip_runtime.h>
#include <cstdint>

// PCM affinity-refined CAM for gfx1250: fused single-pass WMMA + async-LDS staging.
// Workspace layout (needs ~24.1 MB):
//   [0)          ftT : f16, [8][3136][448]  normalized ft, transposed (row j = 448 ch)
//   [22478848)   camx: f16, [8][32][3136]   cam rows 0..20, row21 = ones, 22..31 = 0

typedef __attribute__((ext_vector_type(16))) _Float16 v16h;
typedef __attribute__((ext_vector_type(8)))  _Float16 v8h;
typedef __attribute__((ext_vector_type(8)))  float    v8f;

#define N_BATCH 8
#define C_FT    448
#define HW      3136
#define CAM_R   21
#define KCH     14      // 448 / 32 k-chunks for the Gram GEMM
#define IPAIRS  98      // 196 i-tiles processed in pairs (K=32 for 2nd GEMM)
#define JW      4       // j-tiles (waves) per workgroup sharing one LDS A-tile
#define ATILE_H (32 * C_FT)              // halves per staged A tile (28672 B)
#define EPS_F   1e-5f

// ---------------- prep kernel 1: L2-normalize ft over channels, emit f16 X^T ----
__global__ __launch_bounds__(256) void pcm_normalize_ftT(const float* __restrict__ ft,
                                                         _Float16* __restrict__ ftT) {
  int j = blockIdx.x * blockDim.x + threadIdx.x;
  int b = blockIdx.y;
  if (j >= HW) return;
  const float* p = ft + (size_t)b * C_FT * HW + j;
  float ss = 0.f;
  for (int c = 0; c < C_FT; ++c) { float v = p[(size_t)c * HW]; ss += v * v; }
  float sc = 1.f / (sqrtf(ss) + EPS_F);
  _Float16* q = ftT + ((size_t)b * HW + j) * C_FT;
  for (int c = 0; c < C_FT; ++c) q[c] = (_Float16)(p[(size_t)c * HW] * sc);
}

// ---------------- prep kernel 2: camx = [cam(21) ; ones(1) ; zeros(10)] in f16 ---
__global__ __launch_bounds__(256) void pcm_build_camx(const float* __restrict__ cam,
                                                      _Float16* __restrict__ camx) {
  int idx = blockIdx.x * blockDim.x + threadIdx.x;
  int b = blockIdx.y;
  if (idx >= 32 * HW) return;
  int k = idx / HW, j = idx - k * HW;
  float v;
  if (k < CAM_R)       v = cam[((size_t)b * CAM_R + k) * HW + j];
  else if (k == CAM_R) v = 1.0f;   // ones row -> column sums of relu(G)
  else                 v = 0.0f;
  camx[((size_t)b * 32 + k) * HW + j] = (_Float16)v;
}

// Async 16B global->LDS copy, per-lane addressed, tracked by ASYNCcnt.
__device__ __forceinline__ void async_g2l_b128(unsigned lds_off, const void* gaddr) {
  asm volatile("global_load_async_to_lds_b128 %0, %1, off"
               :: "v"(lds_off), "v"((unsigned long long)(uintptr_t)gaddr)
               : "memory");
}
__device__ __forceinline__ void wait_async0() {
  asm volatile("s_wait_asynccnt 0" ::: "memory");
}

// ---------------- main kernel: 4 waves/WG, each wave owns one 16-col j-tile -----
__global__ __launch_bounds__(128) void pcm_main(const _Float16* __restrict__ ftT,
                                                const _Float16* __restrict__ camx,
                                                float* __restrict__ out) {
  const int tid  = threadIdx.x;
  const int wave = tid >> 5;
  const int lane = tid & 31;
  const int lr   = lane & 15;
  const int hi   = lane >> 4;
  const int jg   = blockIdx.x;               // 0..48
  const int b    = blockIdx.y;               // 0..7
  const int jt   = jg * JW + wave;           // this wave's j-tile, 0..195

  __shared__ alignas(16) _Float16 abuf[2][ATILE_H];   // double-buffered 32x448 A tile

  const _Float16* ftB   = ftT  + (size_t)b * HW * C_FT;
  const _Float16* camxB = camx + (size_t)b * 32 * HW;

  // Per-wave B fragments (columns j0..j0+15), resident in VGPRs for the whole loop.
  // B 32x16 f16 layout: lane = column N, lanes<16 carry K={0..7,16..23}, lanes>=16 K={8..15,24..31}.
  const _Float16* brow = ftB + (size_t)(jt * 16 + lr) * C_FT;
  v16h Bf[KCH];
#pragma unroll
  for (int kc = 0; kc < KCH; ++kc)
    Bf[kc] = __builtin_shufflevector(*(const v8h*)(brow + kc * 32 + 8 * hi), *(const v8h*)(brow + kc * 32 + 16 + 8 * hi), 0,1,2,3,4,5,6,7,8,9,10,11,12,13,14,15);

  // Stage i-pair tile it2 (rows it2*32 .. +32, contiguous 28672 B) into abuf[bufi].
  // 128 threads x 14 lanes-worth of 16B units; each asm op moves 512 B per wave.
  auto stage = [&](int bufi, int it2) {
    const _Float16* src = ftB + (size_t)it2 * ATILE_H;
    unsigned lds0 = (unsigned)(uintptr_t)&abuf[bufi][0];
#pragma unroll
    for (int c = 0; c < 14; ++c) {
      int u = c * 128 + tid;                 // 16-byte unit index
      async_g2l_b128(lds0 + u * 16, src + u * 8);
    }
  };

  stage(0, 0);

  v8f Rlo = {};   // camx rows 0..15  x j-tile accumulator
  v8f Rhi = {};   // camx rows 16..31 (row 21 = colsum) accumulator

  for (int it2 = 0; it2 < IPAIRS; ++it2) {
    wait_async0();          // this wave's copies into buf[it2&1] are done
    __syncthreads();        // every wave's copies done; prev compute finished
    if (it2 + 1 < IPAIRS) stage((it2 + 1) & 1, it2 + 1);   // overlap next DMA

    const _Float16* As = &abuf[it2 & 1][0];
    const int i0 = it2 * 32;

    v8f D0 = {}, D1 = {};
#pragma unroll
    for (int kc = 0; kc < KCH; ++kc) {
      // A 16x32 f16 layout: lane<16 K={0..7,16..23}, lane>=16 K={8..15,24..31}
      const _Float16* ar0 = As + (size_t)lr * C_FT + kc * 32;   // rows i0..i0+15
      v8h a0l = *(const v8h*)(ar0 + 8 * hi);
      v8h a0h = *(const v8h*)(ar0 + 16 + 8 * hi);
      v16h A0 = __builtin_shufflevector(a0l, a0h, 0,1,2,3,4,5,6,7,8,9,10,11,12,13,14,15);
      D0 = __builtin_amdgcn_wmma_f32_16x16x32_f16(false, A0, false, Bf[kc],
                                                  (short)0, D0, false, false);
      const _Float16* ar1 = ar0 + (size_t)16 * C_FT;            // rows i0+16..i0+31
      v8h a1l = *(const v8h*)(ar1 + 8 * hi);
      v8h a1h = *(const v8h*)(ar1 + 16 + 8 * hi);
      v16h A1 = __builtin_shufflevector(a1l, a1h, 0,1,2,3,4,5,6,7,8,9,10,11,12,13,14,15);
      D1 = __builtin_amdgcn_wmma_f32_16x16x32_f16(false, A1, false, Bf[kc],
                                                  (short)0, D1, false, false);
    }

    // ReLU + half-wave exchange: reshape the two f32 D tiles straight into the
    // f16 B fragment (K = 32 i-values x 16 j-cols) of the camx GEMM.
    v16h B2;
#pragma unroll
    for (int v = 0; v < 8; ++v) {
      float r0 = fmaxf(D0[v], 0.f);
      float r1 = fmaxf(D1[v], 0.f);
      float send = hi ? r0 : r1;
      float got  = __shfl_xor(send, 16, 32);
      float e_lo = hi ? got : r0;
      float e_hi = hi ? r1 : got;
      B2[v]     = (_Float16)e_lo;
      B2[v + 8] = (_Float16)e_hi;
    }

    // camx A fragments (K = this 32-wide i-chunk).
    const _Float16* crow0 = camxB + (size_t)lr * HW + i0;       // rows m=0..15
    const _Float16* crow1 = crow0 + (size_t)16 * HW;            // rows m=16..31
    v8h c0l = *(const v8h*)(crow0 + 16 * hi);
    v8h c0h = *(const v8h*)(crow0 + 16 * hi + 8);
    v16h CA0 = __builtin_shufflevector(c0l, c0h, 0,1,2,3,4,5,6,7,8,9,10,11,12,13,14,15);
    v8h c1l = *(const v8h*)(crow1 + 16 * hi);
    v8h c1h = *(const v8h*)(crow1 + 16 * hi + 8);
    v16h CA1 = __builtin_shufflevector(c1l, c1h, 0,1,2,3,4,5,6,7,8,9,10,11,12,13,14,15);

    Rlo = __builtin_amdgcn_wmma_f32_16x16x32_f16(false, CA0, false, B2,
                                                 (short)0, Rlo, false, false);
    Rhi = __builtin_amdgcn_wmma_f32_16x16x32_f16(false, CA1, false, B2,
                                                 (short)0, Rhi, false, false);
  }

  // colsum[j] is camx row 21 -> Rhi element v=5 on lanes<16; broadcast to partner.
  float t  = __shfl_xor(Rhi[5], 16, 32);
  float cs = hi ? t : Rhi[5];
  float inv = 1.f / (cs + EPS_F);

  const int j = jt * 16 + lr;
  float* ob = out + (size_t)b * CAM_R * HW;
#pragma unroll
  for (int v = 0; v < 8; ++v) {                 // k = 0..15
    int k = v + 8 * hi;
    ob[(size_t)k * HW + j] = Rlo[v] * inv;
  }
  if (!hi) {
#pragma unroll
    for (int v = 0; v < 5; ++v)                 // k = 16..20
      ob[(size_t)(16 + v) * HW + j] = Rhi[v] * inv;
  }
}

extern "C" void kernel_launch(void* const* d_in, const int* in_sizes, int n_in,
                              void* d_out, int out_size, void* d_ws, size_t ws_size,
                              hipStream_t stream) {
  const float* cam = (const float*)d_in[0];   // [8][21][56][56]
  const float* ft  = (const float*)d_in[1];   // [8][448][56][56]
  float* out = (float*)d_out;                 // [8][21][56][56]

  _Float16* ftT  = (_Float16*)d_ws;
  _Float16* camx = (_Float16*)((char*)d_ws +
                               (size_t)N_BATCH * HW * C_FT * sizeof(_Float16));

  dim3 g1((HW + 255) / 256, N_BATCH);
  pcm_normalize_ftT<<<g1, 256, 0, stream>>>(ft, ftT);

  dim3 g2((32 * HW + 255) / 256, N_BATCH);
  pcm_build_camx<<<g2, 256, 0, stream>>>(cam, camx);

  dim3 g3(HW / (16 * JW), N_BATCH);           // 49 x 8 workgroups, 4 waves each
  pcm_main<<<g3, 128, 0, stream>>>(ftT, camx, out);
}
